// MultiHeadAttention_56427280335583
// MI455X (gfx1250) — compile-verified, matrix operand order checked by reading, not run on hardware
//
#include <hip/hip_runtime.h>
#include <hip/hip_bf16.h>
#include <math.h>

// MI455X / gfx1250 multi-head attention, bf16 WMMA pipeline, f32 accumulate.
// Stages: cvt/transpose -> QKV GEMM -> flash attention -> output GEMM.
// GEMMs: block-cooperative double-buffered LDS staging using CDNA5 async
// global->LDS copies (GLOBAL_LOAD_ASYNC_TO_LDS_B128, ASYNCcnt). No VGPR
// staging registers -> no spills; one s_wait_asynccnt + barrier per K-step;
// next step's copies are issued right after the barrier so they have a full
// iteration of WMMA work to cover L2 latency. Fragments feed from padded,
// bank-conflict-free LDS rows.

#define T_SEQ  2048
#define C_DIM  1024
#define H_NUM  16
#define D_HEAD 64
#define BATCH  2
#define NTOK   (BATCH * T_SEQ)   // 4096 tokens

#define LDSA_STRIDE 72           // 64 k-elems + 8 pad (144B rows, bank-safe)

typedef __attribute__((ext_vector_type(16))) __bf16 bf16x16;
typedef __attribute__((ext_vector_type(8)))  float  f32x8;

union FragU {
    bf16x16 v;
    uint4   q[2];
};

// round-to-nearest-even f32 -> bf16 bits
__device__ __forceinline__ unsigned short f2bf(float f) {
    unsigned int u = __float_as_uint(f);
    unsigned int r = u + 0x7FFFu + ((u >> 16) & 1u);
    if ((u & 0x7F800000u) == 0x7F800000u) r = u;   // Inf/NaN: truncate
    return (unsigned short)(r >> 16);
}

// Load one 16x32 bf16 WMMA A/B fragment for this lane.
// Caller passes p = row_base + (lane>=16 ? 8 : 0):
// elements [0..7] -> K = base..base+7, elements [8..15] -> K = base+16..base+23
__device__ __forceinline__ bf16x16 load_frag(const unsigned short* p) {
    FragU f;
    f.q[0] = *reinterpret_cast<const uint4*>(p);
    f.q[1] = *reinterpret_cast<const uint4*>(p + 16);
    return f.v;
}

__device__ __forceinline__ f32x8 wmma_bf16(bf16x16 a, bf16x16 b, f32x8 c) {
    return __builtin_amdgcn_wmma_f32_16x16x32_bf16(false, a, false, b, (short)0, c,
                                                   false, false);
}

// CDNA5 async global->LDS copy, 16B per lane, tracked by ASYNCcnt.
// Generic pointers to __shared__ carry the LDS byte offset in their low 32
// bits (ISA 10.2 aperture rules), which is what VDST wants.
__device__ __forceinline__ void async_g2l_b128(const void* gsrc, const void* ldst) {
    asm volatile("global_load_async_to_lds_b128 %0, %1, off"
                 :: "v"((unsigned)(size_t)ldst),
                    "v"((unsigned long long)(size_t)gsrc)
                 : "memory");
}
__device__ __forceinline__ void wait_async0() {
    asm volatile("s_wait_asynccnt 0x0" ::: "memory");
}

// ---------------------------------------------------------------- converters
__global__ void cvt_f32_to_bf16(const float* __restrict__ in,
                                unsigned short* __restrict__ out, int n) {
    int i = blockIdx.x * blockDim.x + threadIdx.x;
    if (i < n) out[i] = f2bf(in[i]);
}

// in: [K][N] f32 row-major  ->  out: [N][K] bf16 (transposed)
__global__ void transpose_to_bf16(const float* __restrict__ in,
                                  unsigned short* __restrict__ out, int K, int N) {
    int i = blockIdx.x * blockDim.x + threadIdx.x;
    if (i < K * N) {
        int n = i / K;
        int k = i - n * K;
        out[i] = f2bf(in[k * N + n]);
    }
}

// ---------------------------------------------------------------- QKV GEMM
// xb[NTOK][C] @ WkqvT[3C][C]^T + b  -> split into k / q(*0.125) / v^T per head
__global__ void __launch_bounds__(256)
gemm_kqv(const unsigned short* __restrict__ xb,
         const unsigned short* __restrict__ wt,
         const float* __restrict__ bias,
         unsigned short* __restrict__ kbuf,
         unsigned short* __restrict__ qbuf,
         unsigned short* __restrict__ vtb) {
    __shared__ __align__(16) unsigned short ldsA[2][128 * LDSA_STRIDE];
    __shared__ __align__(16) unsigned short ldsB[2][64 * LDSA_STRIDE];

    const int lane = threadIdx.x & 31;
    const int wave = threadIdx.x >> 5;
    const int l16  = lane & 15;
    const int half = lane >> 4;
    const int l8   = lane & 7;       // 16B column within a 128B row segment
    const int r4   = lane >> 3;      // 0..3: row group within a copy instr
    const int m0b  = blockIdx.y * 128;
    const int m0   = m0b + wave * 16;
    const int n0   = blockIdx.x * 64;

    // global sources: 4 A-copies (4 rows x 8 lanes each), 2 B-copies per wave
    const unsigned short* ag[4];
    const unsigned short* bg[2];
#pragma unroll
    for (int i = 0; i < 4; ++i)
        ag[i] = xb + (size_t)(m0b + wave * 16 + i * 4 + r4) * C_DIM + l8 * 8;
#pragma unroll
    for (int i = 0; i < 2; ++i)
        bg[i] = wt + (size_t)(n0 + wave * 8 + i * 4 + r4) * C_DIM + l8 * 8;

    f32x8 acc[4] = {};

    // kick off async copies for buffer 0 (kk = 0)
#pragma unroll
    for (int i = 0; i < 4; ++i)
        async_g2l_b128(ag[i],
                       &ldsA[0][(wave * 16 + i * 4 + r4) * LDSA_STRIDE + l8 * 8]);
#pragma unroll
    for (int i = 0; i < 2; ++i)
        async_g2l_b128(bg[i],
                       &ldsB[0][(wave * 8 + i * 4 + r4) * LDSA_STRIDE + l8 * 8]);

    for (int kk = 0; kk < C_DIM; kk += 64) {
        const int p = (kk >> 6) & 1;
        unsigned short* Atile = &ldsA[p][0];
        unsigned short* Btile = &ldsB[p][0];

        wait_async0();        // my copies into buf p complete
        __syncthreads();      // everyone's complete; prior reads of p^1 done

        // issue next K-step's async copies into the other buffer
        if (kk + 64 < C_DIM) {
            const int q = p ^ 1;
#pragma unroll
            for (int i = 0; i < 4; ++i)
                async_g2l_b128(ag[i] + kk + 64,
                    &ldsA[q][(wave * 16 + i * 4 + r4) * LDSA_STRIDE + l8 * 8]);
#pragma unroll
            for (int i = 0; i < 2; ++i)
                async_g2l_b128(bg[i] + kk + 64,
                    &ldsB[q][(wave * 8 + i * 4 + r4) * LDSA_STRIDE + l8 * 8]);
        }

        // fragments from LDS (bank-conflict-free: 144B row stride)
        const unsigned short* Ar = Atile + (wave * 16 + l16) * LDSA_STRIDE + half * 8;
        bf16x16 a0 = load_frag(Ar);
        bf16x16 a1 = load_frag(Ar + 32);
        bf16x16 bf0[4], bf1[4];
#pragma unroll
        for (int s = 0; s < 4; ++s) {
            const unsigned short* Br = Btile + (s * 16 + l16) * LDSA_STRIDE + half * 8;
            bf0[s] = load_frag(Br);
            bf1[s] = load_frag(Br + 32);
        }
#pragma unroll
        for (int s = 0; s < 4; ++s) acc[s] = wmma_bf16(a0, bf0[s], acc[s]);
#pragma unroll
        for (int s = 0; s < 4; ++s) acc[s] = wmma_bf16(a1, bf1[s], acc[s]);
    }

#pragma unroll
    for (int s = 0; s < 4; ++s) {
        const int col = n0 + s * 16 + l16;
        const float bv = bias[col];
        const int sec = col >> 10;          // 0=k, 1=q, 2=v (reference split order)
        const int c   = col & (C_DIM - 1);
        const int h   = c >> 6;
        const int d   = c & (D_HEAD - 1);
#pragma unroll
        for (int g = 0; g < 8; ++g) {
            const int row = m0 + g + half * 8;            // flat token
            const int b   = row >> 11;
            const int t   = row & (T_SEQ - 1);
            const int bh  = b * H_NUM + h;
            const float v = acc[s][g] + bv;
            if (sec == 0) {
                kbuf[((size_t)bh * T_SEQ + t) * D_HEAD + d] = f2bf(v);
            } else if (sec == 1) {
                qbuf[((size_t)bh * T_SEQ + t) * D_HEAD + d] = f2bf(v * 0.125f);
            } else {
                vtb[((size_t)bh * D_HEAD + d) * T_SEQ + t]  = f2bf(v);
            }
        }
    }
}

// ---------------------------------------------------------------- attention
// One wave per 16-row q-tile per (b,h). Flash-attention with online softmax,
// 32 keys per chunk. Chunk schedule: score WMMAs on resident K frags ->
// issue next-chunk K loads + this-chunk V loads -> softmax VALU block hides
// them -> P via LDS -> ctx WMMAs. Two K-buffer sets ping-pong (no copies).
__global__ void __launch_bounds__(256)
attn_kernel(const unsigned short* __restrict__ qb,
            const unsigned short* __restrict__ kb,
            const unsigned short* __restrict__ vt,
            unsigned short* __restrict__ ctxb) {
    __shared__ __align__(16) unsigned short Plds[8][16 * 32];

    const int lane = threadIdx.x & 31;
    const int wave = threadIdx.x >> 5;
    const int l16  = lane & 15;
    const int half = lane >> 4;
    const int w    = blockIdx.x * 8 + wave;   // global wave id, 4096 total
    const int qtile = w & 127;                // T/16
    const int bh    = w >> 7;                 // B*H = 32
    const int q0    = qtile * 16;

    const unsigned short* qrow =
        qb + ((size_t)bh * T_SEQ + q0 + l16) * D_HEAD + half * 8;
    const bf16x16 qf0 = load_frag(qrow);        // d = 0..31
    const bf16x16 qf1 = load_frag(qrow + 32);   // d = 32..63

    f32x8 acc[4] = {};
    float mrow[8], lrow[8];
#pragma unroll
    for (int g = 0; g < 8; ++g) { mrow[g] = -INFINITY; lrow[g] = 0.0f; }

    unsigned short* myP = &Plds[wave][0];
    const unsigned short* kbh = kb + (size_t)bh * T_SEQ * D_HEAD;
    const unsigned short* vbh = vt + (size_t)bh * D_HEAD * T_SEQ;
    const int nchunks = (q0 + 47) >> 5;   // ceil((q0+16)/32)

    // chunk-0 K fragments in buffer set A
    bf16x16 ka0, ka1, ka2, ka3, kb0, kb1, kb2, kb3;
    {
        const unsigned short* kr = kbh + (size_t)l16 * D_HEAD + half * 8;
        ka0 = load_frag(kr);
        ka1 = load_frag(kr + 32);
        ka2 = load_frag(kr + 16 * D_HEAD);
        ka3 = load_frag(kr + 16 * D_HEAD + 32);
    }

    // one 32-key chunk; c* = resident K frags, p* receive next chunk's
    auto chunk = [&](int k0,
                     bf16x16& c0, bf16x16& c1, bf16x16& c2, bf16x16& c3,
                     bf16x16& p0f, bf16x16& p1f, bf16x16& p2f, bf16x16& p3f) {
        // scores for keys [k0, k0+32)
        f32x8 s0 = {}, s1 = {};
        s0 = wmma_bf16(qf0, c0, s0);
        s0 = wmma_bf16(qf1, c1, s0);
        s1 = wmma_bf16(qf0, c2, s1);
        s1 = wmma_bf16(qf1, c3, s1);

        // next-chunk K loads into the other buffer set (tail overrun -> ws)
        const unsigned short* krn =
            kbh + (size_t)(k0 + 32 + l16) * D_HEAD + half * 8;
        p0f = load_frag(krn);
        p1f = load_frag(krn + 32);
        p2f = load_frag(krn + 16 * D_HEAD);
        p3f = load_frag(krn + 16 * D_HEAD + 32);

        // this-chunk V loads (consumed after softmax)
        bf16x16 vf[4];
#pragma unroll
        for (int s = 0; s < 4; ++s)
            vf[s] = load_frag(vbh + (size_t)(s * 16 + l16) * T_SEQ + k0 + half * 8);

        if (k0 + 31 > q0) {   // causal mask — only the diagonal chunk
            const int key0 = k0 + l16;
#pragma unroll
            for (int g = 0; g < 8; ++g) {
                const int row = q0 + g + half * 8;
                if (key0 > row)      s0[g] = -INFINITY;
                if (key0 + 16 > row) s1[g] = -INFINITY;
            }
        }

        // online softmax: rows live across the 16 lanes of each half-wave
        float scale[8];
#pragma unroll
        for (int g = 0; g < 8; ++g) {
            float t = fmaxf(s0[g], s1[g]);
#pragma unroll
            for (int off = 1; off < 16; off <<= 1)
                t = fmaxf(t, __shfl_xor(t, off, 32));
            const float mn = fmaxf(mrow[g], t);
            scale[g] = __expf(mrow[g] - mn);
            const float p0 = __expf(s0[g] - mn);
            const float p1 = __expf(s1[g] - mn);
            s0[g] = p0; s1[g] = p1;
            float rs = p0 + p1;
#pragma unroll
            for (int off = 1; off < 16; off <<= 1)
                rs += __shfl_xor(rs, off, 32);
            lrow[g] = lrow[g] * scale[g] + rs;
            mrow[g] = mn;
        }
#pragma unroll
        for (int d = 0; d < 4; ++d)
#pragma unroll
            for (int g = 0; g < 8; ++g)
                acc[d][g] *= scale[g];

        // P (D-layout) -> LDS row-major 16x32 bf16 -> A-fragment layout
#pragma unroll
        for (int g = 0; g < 8; ++g) {
            const int rl = g + half * 8;
            myP[rl * 32 + l16]      = f2bf(s0[g]);
            myP[rl * 32 + 16 + l16] = f2bf(s1[g]);
        }
        __asm__ volatile("s_wait_dscnt 0x0" ::: "memory");  // LDS RAW, same wave
        const bf16x16 pf = load_frag(myP + l16 * 32 + half * 8);

        // ctx += P(16x32) @ V(32xD)
#pragma unroll
        for (int s = 0; s < 4; ++s)
            acc[s] = wmma_bf16(pf, vf[s], acc[s]);
    };

    for (int kc = 0; kc < nchunks; kc += 2) {
        chunk(kc * 32, ka0, ka1, ka2, ka3, kb0, kb1, kb2, kb3);
        if (kc + 1 < nchunks)
            chunk(kc * 32 + 32, kb0, kb1, kb2, kb3, ka0, ka1, ka2, ka3);
    }

    // normalize and write ctx back as bf16 [tok][C] for the output GEMM
    float inv[8];
#pragma unroll
    for (int g = 0; g < 8; ++g) inv[g] = 1.0f / lrow[g];
    const int b = bh >> 4, h = bh & (H_NUM - 1);
#pragma unroll
    for (int s = 0; s < 4; ++s) {
#pragma unroll
        for (int g = 0; g < 8; ++g) {
            const int tok = b * T_SEQ + q0 + g + half * 8;
            const int col = h * D_HEAD + s * 16 + l16;
            ctxb[(size_t)tok * C_DIM + col] = f2bf(acc[s][g] * inv[g]);
        }
    }
}

// ---------------------------------------------------------------- out proj
__global__ void __launch_bounds__(256)
gemm_out(const unsigned short* __restrict__ ctxb,
         const unsigned short* __restrict__ wt,
         const float* __restrict__ bias,
         float* __restrict__ out) {
    __shared__ __align__(16) unsigned short ldsA[2][128 * LDSA_STRIDE];
    __shared__ __align__(16) unsigned short ldsB[2][64 * LDSA_STRIDE];

    const int lane = threadIdx.x & 31;
    const int wave = threadIdx.x >> 5;
    const int l16  = lane & 15;
    const int half = lane >> 4;
    const int l8   = lane & 7;
    const int r4   = lane >> 3;
    const int m0b  = blockIdx.y * 128;
    const int m0   = m0b + wave * 16;
    const int n0   = blockIdx.x * 64;

    const unsigned short* ag[4];
    const unsigned short* bg[2];
#pragma unroll
    for (int i = 0; i < 4; ++i)
        ag[i] = ctxb + (size_t)(m0b + wave * 16 + i * 4 + r4) * C_DIM + l8 * 8;
#pragma unroll
    for (int i = 0; i < 2; ++i)
        bg[i] = wt + (size_t)(n0 + wave * 8 + i * 4 + r4) * C_DIM + l8 * 8;

    f32x8 acc[4] = {};

#pragma unroll
    for (int i = 0; i < 4; ++i)
        async_g2l_b128(ag[i],
                       &ldsA[0][(wave * 16 + i * 4 + r4) * LDSA_STRIDE + l8 * 8]);
#pragma unroll
    for (int i = 0; i < 2; ++i)
        async_g2l_b128(bg[i],
                       &ldsB[0][(wave * 8 + i * 4 + r4) * LDSA_STRIDE + l8 * 8]);

    for (int kk = 0; kk < C_DIM; kk += 64) {
        const int p = (kk >> 6) & 1;
        unsigned short* Atile = &ldsA[p][0];
        unsigned short* Btile = &ldsB[p][0];

        wait_async0();
        __syncthreads();

        if (kk + 64 < C_DIM) {
            const int q = p ^ 1;
#pragma unroll
            for (int i = 0; i < 4; ++i)
                async_g2l_b128(ag[i] + kk + 64,
                    &ldsA[q][(wave * 16 + i * 4 + r4) * LDSA_STRIDE + l8 * 8]);
#pragma unroll
            for (int i = 0; i < 2; ++i)
                async_g2l_b128(bg[i] + kk + 64,
                    &ldsB[q][(wave * 8 + i * 4 + r4) * LDSA_STRIDE + l8 * 8]);
        }

        const unsigned short* Ar = Atile + (wave * 16 + l16) * LDSA_STRIDE + half * 8;
        bf16x16 a0 = load_frag(Ar);
        bf16x16 a1 = load_frag(Ar + 32);
        bf16x16 bf0[4], bf1[4];
#pragma unroll
        for (int s = 0; s < 4; ++s) {
            const unsigned short* Br = Btile + (s * 16 + l16) * LDSA_STRIDE + half * 8;
            bf0[s] = load_frag(Br);
            bf1[s] = load_frag(Br + 32);
        }
#pragma unroll
        for (int s = 0; s < 4; ++s) acc[s] = wmma_bf16(a0, bf0[s], acc[s]);
#pragma unroll
        for (int s = 0; s < 4; ++s) acc[s] = wmma_bf16(a1, bf1[s], acc[s]);
    }

#pragma unroll
    for (int s = 0; s < 4; ++s) {
        const int col = n0 + s * 16 + l16;
        const float bv = bias[col];
#pragma unroll
        for (int g = 0; g < 8; ++g) {
            const int row = m0 + g + half * 8;
            out[(size_t)row * C_DIM + col] = acc[s][g] + bv;
        }
    }
}

// ---------------------------------------------------------------- launch
extern "C" void kernel_launch(void* const* d_in, const int* in_sizes, int n_in,
                              void* d_out, int out_size, void* d_ws, size_t ws_size,
                              hipStream_t stream) {
    const float* x    = (const float*)d_in[0];   // [B,T,C]
    const float* Wkqv = (const float*)d_in[1];   // [C,3C]
    const float* bkqv = (const float*)d_in[2];   // [3C]
    const float* Wout = (const float*)d_in[3];   // [C,C]
    const float* bout = (const float*)d_in[4];   // [C]
    float* out = (float*)d_out;                  // [B,T,C] f32

    // workspace carve-out (~50 MB + 8 KB tail pad)
    char* ws = (char*)d_ws;
    size_t off = 0;
    auto carve = [&](size_t bytes) -> void* {
        void* p = ws + off;
        off = (off + bytes + 255) & ~(size_t)255;
        return p;
    };
    unsigned short* xb    = (unsigned short*)carve((size_t)NTOK * C_DIM * 2);
    unsigned short* wkqvT = (unsigned short*)carve((size_t)3 * C_DIM * C_DIM * 2);
    unsigned short* woutT = (unsigned short*)carve((size_t)C_DIM * C_DIM * 2);
    unsigned short* qb    = (unsigned short*)carve((size_t)NTOK * C_DIM * 2);
    unsigned short* kbuf  = (unsigned short*)carve((size_t)NTOK * C_DIM * 2);
    unsigned short* vtb   = (unsigned short*)carve((size_t)NTOK * C_DIM * 2);
    unsigned short* ctxb  = (unsigned short*)carve((size_t)NTOK * C_DIM * 2);
    (void)carve(8192);   // pad; K prefetch tail overrun reads the start of vtb
    (void)in_sizes; (void)n_in; (void)out_size; (void)ws_size;

    {   // x -> bf16
        int n = NTOK * C_DIM;
        cvt_f32_to_bf16<<<(n + 255) / 256, 256, 0, stream>>>(x, xb, n);
    }
    {   // W_kqv [C][3C] -> [3C][C] bf16
        int n = C_DIM * 3 * C_DIM;
        transpose_to_bf16<<<(n + 255) / 256, 256, 0, stream>>>(Wkqv, wkqvT,
                                                               C_DIM, 3 * C_DIM);
    }
    {   // W_out [C][C] -> [C][C]^T bf16
        int n = C_DIM * C_DIM;
        transpose_to_bf16<<<(n + 255) / 256, 256, 0, stream>>>(Wout, woutT,
                                                               C_DIM, C_DIM);
    }
    // QKV projection: M=4096, N=3072, K=1024
    gemm_kqv<<<dim3(48, 32), 256, 0, stream>>>(xb, wkqvT, bkqv, kbuf, qb, vtb);
    // attention: 4096 q-tiles, 8 waves/block
    attn_kernel<<<512, 256, 0, stream>>>(qb, kbuf, vtb, ctxb);
    // output projection: M=4096, N=1024, K=1024
    gemm_out<<<dim3(16, 32), 256, 0, stream>>>(ctxb, woutT, bout, out);
}
